// SalGATVAE_55430847922552
// MI455X (gfx1250) — matrix operand order repaired and checked by reading, not compiled, not run on hardware
//
#include <hip/hip_runtime.h>
#include <hip/hip_bf16.h>

#define LRELU_ALPHA 0.2f

typedef __attribute__((ext_vector_type(16))) __bf16 v16bf;
typedef __attribute__((ext_vector_type(8)))  float  v8f;
typedef __attribute__((ext_vector_type(4)))  int    v4i;

#if __has_builtin(__builtin_amdgcn_global_load_async_to_lds_b128) && \
    __has_builtin(__builtin_amdgcn_s_wait_asynccnt)
#define HAVE_GFX1250_ASYNC 1
#endif

__device__ __forceinline__ unsigned short f2bf(float f) {
  unsigned u = __float_as_uint(f);
  u += 0x7fffu + ((u >> 16) & 1u);   // round-to-nearest-even
  return (unsigned short)(u >> 16);
}

__device__ __forceinline__ float fast_sigmoid(float x) {
  return __builtin_amdgcn_rcpf(1.f + __expf(-x));
}

// workspace layout (float indices)
#define OFF_V1  0
#define OFF_V2  128
#define OFF_C   256
#define OFF_S1  512
#define OFF_S2  (512 + 16384)
#define OFF_XBF 33280          // start of bf16 copy of x (cast to ushort*)

// ---------------------------------------------------------------------------
// Kernel 1: rank-1 collapse.  v1[w] = sum_e lin_w[e,w]*a1[e]; c1 = lin_b . a1
// ---------------------------------------------------------------------------
__global__ void prep_vec_kernel(const float* __restrict__ lin_w,
                                const float* __restrict__ lin_b,
                                const float* __restrict__ a,
                                float* __restrict__ ws) {
  __shared__ float red1[128], red2[128];
  const int w = threadIdx.x;                 // 0..127
  float acc1 = 0.f, acc2 = 0.f;
  for (int e = 0; e < 128; ++e) {
    float lw = lin_w[e * 128 + w];
    acc1 = fmaf(lw, a[e],       acc1);
    acc2 = fmaf(lw, a[128 + e], acc2);
  }
  ws[OFF_V1 + w] = acc1;
  ws[OFF_V2 + w] = acc2;
  red1[w] = lin_b[w] * a[w];
  red2[w] = lin_b[w] * a[128 + w];
  __syncthreads();
  for (int s = 64; s > 0; s >>= 1) {
    if (w < s) { red1[w] += red1[w + s]; red2[w] += red2[w + s]; }
    __syncthreads();
  }
  if (w == 0) { ws[OFF_C + 0] = red1[0]; ws[OFF_C + 1] = red2[0]; }
}

// ---------------------------------------------------------------------------
// Kernel 2: s1[b,k] = x[b,:,k].v1 + c1 ; s2 likewise ; xbf = bf16(x)
// ---------------------------------------------------------------------------
__global__ void prep_sx_kernel(const float* __restrict__ x,
                               float* __restrict__ ws) {
  __shared__ float v1s[128], v2s[128];
  const int tid = threadIdx.x;
  if (tid < 128) { v1s[tid] = ws[OFF_V1 + tid]; v2s[tid] = ws[OFF_V2 + tid]; }
  __syncthreads();
  const int t = blockIdx.x * 256 + tid;      // 0..16383
  const int b = t >> 11;
  const int k = t & 2047;
  unsigned short* xbf = (unsigned short*)(ws + OFF_XBF);
  float a1 = ws[OFF_C + 0];
  float a2 = ws[OFF_C + 1];
  const size_t basek = (size_t)b * 128 * 2048 + (size_t)k;
  for (int w = 0; w < 128; ++w) {
    float xv = x[basek + (size_t)w * 2048];
    a1 = fmaf(xv, v1s[w], a1);
    a2 = fmaf(xv, v2s[w], a2);
    xbf[basek + (size_t)w * 2048] = f2bf(xv);
  }
  ws[OFF_S1 + b * 2048 + k] = a1;
  ws[OFF_S2 + b * 2048 + k] = a2;
}

// ---------------------------------------------------------------------------
// Kernel 3: fused softmax( lrelu(s1+s2)+bias ) @ xp  -> sigmoid -> transposed
// One wave = one 16-row tile; 4 waves/WG; grid = 8 batches * 32 row-blocks.
// ---------------------------------------------------------------------------
__global__ void __launch_bounds__(128) gat_main_kernel(
    const float* __restrict__ bias,
    const float* __restrict__ ws,
    float* __restrict__ out) {
  __shared__ float s2s[2048];
  const int tid  = threadIdx.x;
  const int lane = tid & 31;
  const int wave = tid >> 5;
  const int b    = blockIdx.x >> 5;
  const int rb   = blockIdx.x & 31;
  const int i0   = rb * 64 + wave * 16;

  const float* s1g = ws + OFF_S1 + b * 2048;
  const float* s2g = ws + OFF_S2 + b * 2048;
  const unsigned short* xbf =
      (const unsigned short*)(ws + OFF_XBF) + (size_t)b * 128 * 2048;

#ifdef HAVE_GFX1250_ASYNC
  for (int t = tid * 4; t < 2048; t += 512) {
    __builtin_amdgcn_global_load_async_to_lds_b128(
        (v4i*)(s2g + t), (v4i*)(&s2s[t]), 0, 0);
  }
  __builtin_amdgcn_s_wait_asynccnt(0);
  __syncthreads();
#else
  for (int t = tid; t < 2048; t += 128) s2s[t] = s2g[t];
  __syncthreads();
#endif

  // ---- pass 1: per-row safe-softmax stats, 4 independent online streams ----
  float rmax = 0.f, rsum = 1.f;
  for (int m = 0; m < 16; ++m) {
    const int i = i0 + m;
    const float s1v = s1g[i];
    const float* brow = bias + (size_t)i * 2048;
    float mx[4] = {-3.4e38f, -3.4e38f, -3.4e38f, -3.4e38f};
    float sm[4] = {0.f, 0.f, 0.f, 0.f};
    for (int j = lane; j < 2048; j += 128) {
      if (j + 512 < 2048) __builtin_prefetch(brow + j + 512, 0, 3);
      #pragma unroll
      for (int q = 0; q < 4; ++q) {
        const int jj = j + q * 32;
        float v = s1v + s2s[jj];
        v = v > 0.f ? v : LRELU_ALPHA * v;
        v += brow[jj];
        float nm = fmaxf(mx[q], v);
        sm[q] = sm[q] * __expf(mx[q] - nm) + __expf(v - nm);
        mx[q] = nm;
      }
    }
    // merge the 4 streams
    #pragma unroll
    for (int q = 1; q < 4; ++q) {
      float nm = fmaxf(mx[0], mx[q]);
      sm[0] = sm[0] * __expf(mx[0] - nm) + sm[q] * __expf(mx[q] - nm);
      mx[0] = nm;
    }
    float mxx = mx[0], smm = sm[0];
    #pragma unroll
    for (int off = 16; off >= 1; off >>= 1) {
      float om = __shfl_xor(mxx, off, 32);
      float os = __shfl_xor(smm, off, 32);
      float nm = fmaxf(mxx, om);
      smm = smm * __expf(mxx - nm) + os * __expf(om - nm);
      mxx = nm;
    }
    if ((lane & 15) == m) { rmax = mxx; rsum = smm; }   // keep my row's stats
  }
  const float lse = rmax + __logf(rsum);   // p = exp(v - lse)

  // ---- pass 2: software-pipelined  P-tile build + WMMA ----
  const int hf   = lane >> 4;     // K-group half
  const int mrow = lane & 15;     // my M row / my N col
  const int irow = i0 + mrow;
  const float s1v = s1g[irow];
  const float* brow = bias + (size_t)irow * 2048;

  v8f acc[8];
  v8f zero = {0.f, 0.f, 0.f, 0.f, 0.f, 0.f, 0.f, 0.f};
  #pragma unroll
  for (int n = 0; n < 8; ++n) acc[n] = zero;

  union AU { v16bf v; __bf16 h[16]; };
  AU aA, aB;
  v16bf btA[8], btB[8];

  // B tiles: x[b][w][j] bf16 row-major == desired 32x16 B layout
  // lane: N = mrow (w within tile), K = hf*16 + e (j contiguous) -> 32B load
  auto loadB = [&](int jc, v16bf* bt) {
    const unsigned short* bbase = xbf + jc + hf * 16;
    #pragma unroll
    for (int n = 0; n < 8; ++n)
      bt[n] = *(const v16bf*)(bbase + (size_t)(n * 16 + mrow) * 2048);
  };
  auto makeA = [&](int jc, AU& au) {
    if (jc + 256 < 2048) __builtin_prefetch(brow + jc + 256 + hf * 8, 0, 3);
    const int j0 = jc + hf * 16;       // A elements 0..7  -> K = hf*16 + e
    const int j1 = jc + hf * 16 + 8;   // A elements 8..15 -> K = hf*16 + 8 + e
    float sv[8], bv[8], pv[16];
    ((float4*)sv)[0] = *(const float4*)(s2s + j0);
    ((float4*)sv)[1] = *(const float4*)(s2s + j0 + 4);
    ((float4*)bv)[0] = *(const float4*)(brow + j0);
    ((float4*)bv)[1] = *(const float4*)(brow + j0 + 4);
    #pragma unroll
    for (int e = 0; e < 8; ++e) {
      float v = s1v + sv[e];
      v = v > 0.f ? v : LRELU_ALPHA * v;
      pv[e] = __expf(v + bv[e] - lse);
    }
    ((float4*)sv)[0] = *(const float4*)(s2s + j1);
    ((float4*)sv)[1] = *(const float4*)(s2s + j1 + 4);
    ((float4*)bv)[0] = *(const float4*)(brow + j1);
    ((float4*)bv)[1] = *(const float4*)(brow + j1 + 4);
    #pragma unroll
    for (int e = 0; e < 8; ++e) {
      float v = s1v + sv[e];
      v = v > 0.f ? v : LRELU_ALPHA * v;
      pv[8 + e] = __expf(v + bv[e] - lse);
    }
    #pragma unroll
    for (int e = 0; e < 16; ++e) au.h[e] = (__bf16)pv[e];  // v_cvt_pk_bf16_f32
  };
  auto wmma8 = [&](const AU& au, const v16bf* bt) {
    #pragma unroll
    for (int n = 0; n < 8; ++n)
      acc[n] = __builtin_amdgcn_wmma_f32_16x16x32_bf16(
          false, au.v, false, bt[n], (short)0, acc[n], false, false);
  };

  loadB(0, btA);
  makeA(0, aA);
  for (int jc = 0; jc < 1984; jc += 64) {
    loadB(jc + 32, btB);
    makeA(jc + 32, aB);
    wmma8(aA, btA);                 // chunk jc
    loadB(jc + 64, btA);
    makeA(jc + 64, aA);
    wmma8(aB, btB);                 // chunk jc+32
  }
  // tail: chunks 1984 (in A-side) and 2016
  loadB(2016, btB);
  makeA(2016, aB);
  wmma8(aA, btA);
  wmma8(aB, btB);

  // ---- sigmoid + transposed store: out[b][w][i], rows contiguous in VGPR r ----
  const size_t outb = (size_t)b * 128 * 2048;
  #pragma unroll
  for (int n = 0; n < 8; ++n) {
    const int w = n * 16 + mrow;
    float* orow = out + outb + (size_t)w * 2048 + i0 + hf * 8;
    float4 o0, o1;
    o0.x = fast_sigmoid(acc[n][0]);
    o0.y = fast_sigmoid(acc[n][1]);
    o0.z = fast_sigmoid(acc[n][2]);
    o0.w = fast_sigmoid(acc[n][3]);
    o1.x = fast_sigmoid(acc[n][4]);
    o1.y = fast_sigmoid(acc[n][5]);
    o1.z = fast_sigmoid(acc[n][6]);
    o1.w = fast_sigmoid(acc[n][7]);
    *(float4*)(orow)     = o0;
    *(float4*)(orow + 4) = o1;
  }
}

// ---------------------------------------------------------------------------
extern "C" void kernel_launch(void* const* d_in, const int* in_sizes, int n_in,
                              void* d_out, int out_size, void* d_ws, size_t ws_size,
                              hipStream_t stream) {
  const float* x     = (const float*)d_in[0];   // (8,128,2048)
  const float* lin_w = (const float*)d_in[1];   // (128,128)
  const float* lin_b = (const float*)d_in[2];   // (128,)
  const float* a     = (const float*)d_in[3];   // (256,1)
  const float* bias  = (const float*)d_in[4];   // (2048,2048)
  float* out = (float*)d_out;                   // (8,128,2048)
  float* ws  = (float*)d_ws;

  prep_vec_kernel<<<dim3(1),   dim3(128), 0, stream>>>(lin_w, lin_b, a, ws);
  prep_sx_kernel <<<dim3(64),  dim3(256), 0, stream>>>(x, ws);
  gat_main_kernel<<<dim3(256), dim3(128), 0, stream>>>(bias, ws, out);
}
